// SwitchTransformerEncoderLayer_16690242913248
// MI455X (gfx1250) — matrix operand order repaired and checked by reading, not compiled, not run on hardware
//
#include <hip/hip_runtime.h>

// Problem constants (match reference)
#define S_   2048
#define B_   2
#define D_   768
#define H_   12
#define F_   3072
#define E_   8
#define DH_  64
#define T_   4096      // S*B
#define CAP_ 1024      // 2*T/E
#define TD3_ 2304      // 3*D

typedef _Float16 f16;
typedef _Float16 v16h __attribute__((ext_vector_type(16)));
typedef float    v8f  __attribute__((ext_vector_type(8)));
typedef unsigned int u32x4 __attribute__((ext_vector_type(4)));

union FragU { v16h h; u32x4 u[2]; };

// A-fragment (16x32 f16): lane<16 -> K 0..15 contiguous; lane>=16 -> K 16..31.
// caller passes p already offset by half*16; chunks at +0 and +8 halves.
__device__ inline v16h loadA32(const f16* p) {
    FragU f;
    f.u[0] = *(const u32x4*)(p);
    f.u[1] = *(const u32x4*)(p + 8);
    return f.h;
}
// B-fragment (32x16 f16): lane<16 -> K 0..15 contiguous; lane>=16 -> K 16..31.
// caller passes p already offset by half*16; chunks at +0 and +8 halves.
__device__ inline v16h loadB32(const f16* p) {
    FragU f;
    f.u[0] = *(const u32x4*)(p);
    f.u[1] = *(const u32x4*)(p + 8);
    return f.h;
}

__device__ inline float redmax16(float v) {
#pragma unroll
    for (int m = 1; m < 16; m <<= 1) v = fmaxf(v, __shfl_xor(v, m, 32));
    return v;
}
__device__ inline float redsum16(float v) {
#pragma unroll
    for (int m = 1; m < 16; m <<= 1) v += __shfl_xor(v, m, 32);
    return v;
}
__device__ inline float redsum32(float v) {
#pragma unroll
    for (int m = 1; m < 32; m <<= 1) v += __shfl_xor(v, m, 32);
    return v;
}

// ---------------------------------------------------------------------------
// Generic "NT" WMMA GEMM: C[M,N] = A[M,K](f16,row-major) x B[N,K](f16,row-major)^T
// + bias[N], optional relu, optional batch (blockIdx.z) with element strides.
// block = (32,8): 8 waves, each wave computes a 16(M) x 64(N) strip.
// Requires M%128==0, N%64==0, K%32==0 (true for all uses here).
// ---------------------------------------------------------------------------
template <typename TOUT, bool RELU>
__global__ __launch_bounds__(256) void gemm_nt(
    const f16* __restrict__ A, const f16* __restrict__ B,
    const float* __restrict__ bias, TOUT* __restrict__ C,
    int M, int N, int K,
    long long saA, long long saB, long long saBias, long long saC)
{
    const long long z = blockIdx.z;
    A += z * saA;
    B += z * saB;
    C += z * saC;
    const float* bi = bias ? (bias + z * saBias) : nullptr;

    const int lane = threadIdx.x;
    const int l16  = lane & 15;
    const int half = lane >> 4;
    const int mtile = blockIdx.y * 8 + threadIdx.y;
    const int n0    = blockIdx.x * 64;

    const f16* arow = A + (size_t)(mtile * 16 + l16) * K + half * 16;
    const f16* b0   = B + (size_t)(n0 + l16) * K + half * 16;
    const size_t bstep = (size_t)16 * K;

    v8f acc[4];
#pragma unroll
    for (int j = 0; j < 4; j++) {
#pragma unroll
        for (int r = 0; r < 8; r++) acc[j][r] = 0.f;
    }

    for (int k = 0; k < K; k += 32) {
        // prefetch ~4 k-steps ahead (gfx1250 global_prefetch_b8)
        if (k + 128 < K) {
            __builtin_prefetch((const void*)(arow + k + 128), 0, 1);
#pragma unroll
            for (int j = 0; j < 4; j++)
                __builtin_prefetch((const void*)(b0 + (size_t)j * bstep + k + 128), 0, 1);
        }
        v16h a = loadA32(arow + k);
#pragma unroll
        for (int j = 0; j < 4; j++) {
            v16h b = loadB32(b0 + (size_t)j * bstep + k);
            acc[j] = __builtin_amdgcn_wmma_f32_16x16x32_f16(
                false, a, false, b, (short)0, acc[j], false, false);
        }
    }

#pragma unroll
    for (int j = 0; j < 4; j++) {
        int col = n0 + j * 16 + l16;
        float bv = bi ? bi[col] : 0.f;
#pragma unroll
        for (int r = 0; r < 8; r++) {
            int row = mtile * 16 + r + 8 * half;
            float v = acc[j][r] + bv;
            if (RELU) v = v > 0.f ? v : 0.f;
            C[(size_t)row * N + col] = (TOUT)v;
        }
    }
}

// ---------------------------------------------------------------------------
// Flash attention, one wave per 16-query tile per (b,h).
// Qh,Kh: [B*H, S, 64] f16 (Q pre-scaled by 1/8). Vt: [B*H, 64, S] f16.
// ctx out: [T, D] f16 (token-major, t = s*B + b, feature = h*64 + d).
// ---------------------------------------------------------------------------
__global__ __launch_bounds__(256) void flash_attn(
    const f16* __restrict__ Qh, const f16* __restrict__ Kh,
    const f16* __restrict__ Vt, f16* __restrict__ ctx)
{
    __shared__ __align__(16) f16 psh[8][16 * 32];

    const int bh  = blockIdx.x;
    const int b   = bh / H_;
    const int h   = bh % H_;
    const int qt  = blockIdx.y * 8 + threadIdx.y;
    const int lane = threadIdx.x;
    const int l16  = lane & 15;
    const int half = lane >> 4;

    const f16* Qp = Qh + ((size_t)bh * S_ + qt * 16 + l16) * DH_ + half * 16;
    v16h q0 = loadA32(Qp);
    v16h q1 = loadA32(Qp + 32);

    const f16* Kb = Kh + (size_t)bh * S_ * DH_;
    const f16* Vb = Vt + (size_t)bh * DH_ * S_;

    float rm[8], rl[8];
    v8f acc[4];
#pragma unroll
    for (int r = 0; r < 8; r++) { rm[r] = -3.0e30f; rl[r] = 0.f; }
#pragma unroll
    for (int j = 0; j < 4; j++) {
#pragma unroll
        for (int r = 0; r < 8; r++) acc[j][r] = 0.f;
    }

    f16* myp = &psh[threadIdx.y][0];

    for (int kt = 0; kt < S_; kt += 32) {
        const f16* kp0 = Kb + (size_t)(kt + l16) * DH_ + half * 16;        // keys kt..kt+15
        const f16* kp1 = Kb + (size_t)(kt + 16 + l16) * DH_ + half * 16;   // keys kt+16..kt+31

        // prefetch next key block's K rows and V rows (global_prefetch_b8)
        if (kt + 32 < S_) {
            __builtin_prefetch((const void*)(kp0 + 32 * DH_), 0, 1);
            __builtin_prefetch((const void*)(kp1 + 32 * DH_), 0, 1);
            __builtin_prefetch((const void*)(Vb + (size_t)l16 * S_ + kt + 32 + half * 16), 0, 1);
        }

        v8f s0, s1;
#pragma unroll
        for (int r = 0; r < 8; r++) { s0[r] = 0.f; s1[r] = 0.f; }
        s0 = __builtin_amdgcn_wmma_f32_16x16x32_f16(false, q0, false, loadB32(kp0),      (short)0, s0, false, false);
        s0 = __builtin_amdgcn_wmma_f32_16x16x32_f16(false, q1, false, loadB32(kp0 + 32), (short)0, s0, false, false);
        s1 = __builtin_amdgcn_wmma_f32_16x16x32_f16(false, q0, false, loadB32(kp1),      (short)0, s1, false, false);
        s1 = __builtin_amdgcn_wmma_f32_16x16x32_f16(false, q1, false, loadB32(kp1 + 32), (short)0, s1, false, false);

        float corr[8];
#pragma unroll
        for (int r = 0; r < 8; r++) {
            float mx = redmax16(fmaxf(s0[r], s1[r]));
            float nm = fmaxf(rm[r], mx);
            corr[r] = __expf(rm[r] - nm);
            float p0 = __expf(s0[r] - nm);
            float p1 = __expf(s1[r] - nm);
            float rs = redsum16(p0 + p1);
            rl[r] = rl[r] * corr[r] + rs;
            rm[r] = nm;
            int m = r + 8 * half;                  // C-layout row owned by this lane
            myp[m * 32 + l16]      = (f16)p0;      // key-local = l16
            myp[m * 32 + 16 + l16] = (f16)p1;      // key-local = 16 + l16
        }
#pragma unroll
        for (int j = 0; j < 4; j++) {
#pragma unroll
            for (int r = 0; r < 8; r++) acc[j][r] *= corr[r];
        }

        asm volatile("s_wait_dscnt 0" ::: "memory");

        // P as A-fragment (16 queries x 32 keys), read back from LDS row-major
        FragU pa;
        pa.u[0] = *(const u32x4*)(myp + l16 * 32 + half * 16);
        pa.u[1] = *(const u32x4*)(myp + l16 * 32 + half * 16 + 8);

#pragma unroll
        for (int j = 0; j < 4; j++) {
            const f16* vp = Vb + (size_t)(j * 16 + l16) * S_ + kt + half * 16;
            acc[j] = __builtin_amdgcn_wmma_f32_16x16x32_f16(
                false, pa.h, false, loadB32(vp), (short)0, acc[j], false, false);
        }
        asm volatile("s_wait_dscnt 0" ::: "memory");
    }

#pragma unroll
    for (int j = 0; j < 4; j++) {
#pragma unroll
        for (int r = 0; r < 8; r++) {
            int m = r + 8 * half;
            int srow = qt * 16 + m;
            int d = j * 16 + l16;
            float v = acc[j][r] / rl[r];
            ctx[((size_t)srow * B_ + b) * D_ + h * DH_ + d] = (f16)v;
        }
    }
}

// ---------------------------------------------------------------------------
// Small helper kernels
// ---------------------------------------------------------------------------
__global__ void cvt_f16(const float* __restrict__ in, f16* __restrict__ out, int n) {
    int i = blockIdx.x * 256 + threadIdx.x;
    if (i < n) out[i] = (f16)in[i];
}

__global__ void zero16(f16* __restrict__ p, int n) {
    int i = blockIdx.x * 256 + threadIdx.x;
    if (i < n) p[i] = (f16)0.f;
}

// w1 [E,D,F] -> w1t [E,F,D]   (K-major over D for the h GEMM)
__global__ void tr_w1(const float* __restrict__ w1, f16* __restrict__ w1t, int n) {
    int i = blockIdx.x * 256 + threadIdx.x;
    if (i >= n) return;
    int e = i / (D_ * F_);
    int r = i % (D_ * F_);
    int d = r / F_;
    int f = r % F_;
    w1t[((size_t)e * F_ + f) * D_ + d] = (f16)w1[i];
}

// w2 [E,F,D] -> w2t [E,D,F]   (K-major over F for the y GEMM)
__global__ void tr_w2(const float* __restrict__ w2, f16* __restrict__ w2t, int n) {
    int i = blockIdx.x * 256 + threadIdx.x;
    if (i >= n) return;
    int e = i / (F_ * D_);
    int r = i % (F_ * D_);
    int f = r / D_;
    int d = r % D_;
    w2t[((size_t)e * D_ + d) * F_ + f] = (f16)w2[i];
}

// qkv f32 [T,3D] -> Qh,Kh [B*H,S,64] f16 (Q scaled 1/8), Vt [B*H,64,S] f16
__global__ void head_split(const float* __restrict__ qkv,
                           f16* __restrict__ Qh, f16* __restrict__ Kh, f16* __restrict__ Vt) {
    int i = blockIdx.x * 256 + threadIdx.x;
    if (i >= T_ * D_) return;
    int t = i / D_, hd = i % D_;
    int s = t / B_, b = t % B_;
    int h = hd / DH_, d = hd % DH_;
    size_t base = (size_t)t * TD3_;
    size_t hb = (size_t)(b * H_ + h);
    Qh[(hb * S_ + s) * DH_ + d] = (f16)(qkv[base + hd] * 0.125f);
    Kh[(hb * S_ + s) * DH_ + d] = (f16)(qkv[base + D_ + hd]);
    Vt[(hb * DH_ + d) * S_ + s] = (f16)(qkv[base + 2 * D_ + hd]);
}

// x = LN(a + r) ; also emits f16 copy of x
__global__ __launch_bounds__(256) void ln_add(
    const float* __restrict__ a, const float* __restrict__ r,
    const float* __restrict__ g, const float* __restrict__ beta,
    float* __restrict__ out, f16* __restrict__ out16)
{
    int t = blockIdx.x * 8 + threadIdx.y;
    int lane = threadIdx.x;
    float v[24];
    float s = 0.f, s2 = 0.f;
#pragma unroll
    for (int i = 0; i < 24; i++) {
        int d = lane + i * 32;
        float x = a[(size_t)t * D_ + d] + r[(size_t)t * D_ + d];
        v[i] = x; s += x; s2 += x * x;
    }
    s = redsum32(s); s2 = redsum32(s2);
    float mean = s * (1.f / D_);
    float var = s2 * (1.f / D_) - mean * mean;
    float rstd = rsqrtf(var + 1e-5f);
#pragma unroll
    for (int i = 0; i < 24; i++) {
        int d = lane + i * 32;
        float o = (v[i] - mean) * rstd * g[d] + beta[d];
        out[(size_t)t * D_ + d] = o;
        out16[(size_t)t * D_ + d] = (f16)o;
    }
}

__global__ void router_kernel(const float* __restrict__ x,
                              const float* __restrict__ rw, const float* __restrict__ rb,
                              float* __restrict__ probs, int* __restrict__ eidx,
                              float* __restrict__ gate)
{
    int t = blockIdx.x * blockDim.x + threadIdx.x;
    if (t >= T_) return;
    const float* xp = x + (size_t)t * D_;
    float lg[E_];
#pragma unroll
    for (int e = 0; e < E_; e++) {
        float s = rb[e];
        const float* w = rw + (size_t)e * D_;
        for (int d = 0; d < D_; d++) s += xp[d] * w[d];
        lg[e] = s;
    }
    float mx = lg[0];
#pragma unroll
    for (int e = 1; e < E_; e++) mx = fmaxf(mx, lg[e]);
    float sum = 0.f;
#pragma unroll
    for (int e = 0; e < E_; e++) { lg[e] = __expf(lg[e] - mx); sum += lg[e]; }
    int bi = 0; float bp = -1.f;
#pragma unroll
    for (int e = 0; e < E_; e++) {
        float p = lg[e] / sum;
        probs[(size_t)t * E_ + e] = p;
        if (p > bp) { bp = p; bi = e; }   // first max wins, matches argmax
    }
    eidx[t] = bi;
    gate[t] = bp;
}

// position of each token in its expert queue (deterministic sequential scan per expert)
__global__ void expert_pos(const int* __restrict__ eidx, int* __restrict__ pos) {
    int e = threadIdx.x;
    if (e >= E_) return;
    int c = 0;
    for (int t = 0; t < T_; t++) {
        if (eidx[t] == e) { pos[t] = c; c++; }
    }
}

__global__ void scatter_buf(const f16* __restrict__ x16, const int* __restrict__ eidx,
                            const int* __restrict__ pos, f16* __restrict__ buf) {
    int t = blockIdx.x;
    int p = pos[t];
    if (p >= CAP_) return;
    int e = eidx[t];
    const f16* src = x16 + (size_t)t * D_;
    f16* dst = buf + ((size_t)e * CAP_ + p) * D_;
    for (int d = threadIdx.x; d < D_; d += 256) dst[d] = src[d];
}

// out = LN(x + gate * y[eidx,pos])   (dropped tokens contribute 0)
__global__ __launch_bounds__(256) void ln2_final(
    const float* __restrict__ x, const float* __restrict__ y,
    const int* __restrict__ eidx, const int* __restrict__ pos,
    const float* __restrict__ gate,
    const float* __restrict__ g, const float* __restrict__ beta,
    float* __restrict__ out)
{
    int t = blockIdx.x * 8 + threadIdx.y;
    int lane = threadIdx.x;
    int e = eidx[t];
    int p = pos[t];
    float gt = gate[t];
    bool keep = p < CAP_;
    const float* yrow = y + ((size_t)e * CAP_ + (keep ? p : 0)) * D_;
    float v[24];
    float s = 0.f, s2 = 0.f;
#pragma unroll
    for (int i = 0; i < 24; i++) {
        int d = lane + i * 32;
        float xv = x[(size_t)t * D_ + d];
        if (keep) xv += yrow[d] * gt;
        v[i] = xv; s += xv; s2 += xv * xv;
    }
    s = redsum32(s); s2 = redsum32(s2);
    float mean = s * (1.f / D_);
    float var = s2 * (1.f / D_) - mean * mean;
    float rstd = rsqrtf(var + 1e-5f);
#pragma unroll
    for (int i = 0; i < 24; i++) {
        int d = lane + i * 32;
        out[(size_t)t * D_ + d] = (v[i] - mean) * rstd * g[d] + beta[d];
    }
}

// load-balance loss, deterministic tree reduction, single block
__global__ __launch_bounds__(256) void moe_loss(const float* __restrict__ probs,
                                                const int* __restrict__ eidx,
                                                float* __restrict__ loss_out)
{
    __shared__ float sred[256];
    int tid = threadIdx.x;
    float fc[E_], ps[E_];
#pragma unroll
    for (int e = 0; e < E_; e++) { fc[e] = 0.f; ps[e] = 0.f; }
    for (int t = tid; t < T_; t += 256) {
        fc[eidx[t]] += 1.f;
#pragma unroll
        for (int e = 0; e < E_; e++) ps[e] += probs[(size_t)t * E_ + e];
    }
    float loss = 0.f;
    for (int e = 0; e < E_; e++) {
        sred[tid] = fc[e]; __syncthreads();
        for (int s = 128; s > 0; s >>= 1) { if (tid < s) sred[tid] += sred[tid + s]; __syncthreads(); }
        float fe = sred[0]; __syncthreads();
        sred[tid] = ps[e]; __syncthreads();
        for (int s = 128; s > 0; s >>= 1) { if (tid < s) sred[tid] += sred[tid + s]; __syncthreads(); }
        float pe = sred[0]; __syncthreads();
        loss += (fe / (float)T_) * (pe / (float)T_);
    }
    if (tid == 0) loss_out[0] = (float)E_ * loss;
}

// ---------------------------------------------------------------------------
extern "C" void kernel_launch(void* const* d_in, const int* in_sizes, int n_in,
                              void* d_out, int out_size, void* d_ws, size_t ws_size,
                              hipStream_t stream)
{
    (void)in_sizes; (void)n_in; (void)out_size; (void)ws_size;

    const float* src  = (const float*)d_in[0];
    const float* inw  = (const float*)d_in[1];
    const float* inb  = (const float*)d_in[2];
    const float* outw = (const float*)d_in[3];
    const float* outb = (const float*)d_in[4];
    const float* n1g  = (const float*)d_in[5];
    const float* n1b  = (const float*)d_in[6];
    const float* rw   = (const float*)d_in[7];
    const float* rb   = (const float*)d_in[8];
    const float* w1   = (const float*)d_in[9];
    const float* b1   = (const float*)d_in[10];
    const float* w2   = (const float*)d_in[11];
    const float* b2   = (const float*)d_in[12];
    const float* n2g  = (const float*)d_in[13];
    const float* n2b  = (const float*)d_in[14];

    char* ws = (char*)d_ws;
    size_t off = 0;
    auto alloc = [&](size_t bytes) -> char* {
        char* p = ws + off;
        off += (bytes + 255) & ~(size_t)255;
        return p;
    };

    f16*   src16  = (f16*)  alloc((size_t)T_ * D_ * 2);
    f16*   inw16  = (f16*)  alloc((size_t)TD3_ * D_ * 2);
    f16*   outw16 = (f16*)  alloc((size_t)D_ * D_ * 2);
    float* qkv    = (float*)alloc((size_t)T_ * TD3_ * 4);
    f16*   Qh     = (f16*)  alloc((size_t)B_ * H_ * S_ * DH_ * 2);
    f16*   Kh     = (f16*)  alloc((size_t)B_ * H_ * S_ * DH_ * 2);
    f16*   Vt     = (f16*)  alloc((size_t)B_ * H_ * S_ * DH_ * 2);
    f16*   ctx16  = (f16*)  alloc((size_t)T_ * D_ * 2);
    float* attn   = (float*)alloc((size_t)T_ * D_ * 4);
    float* x      = (float*)alloc((size_t)T_ * D_ * 4);
    f16*   x16    = (f16*)  alloc((size_t)T_ * D_ * 2);
    float* probs  = (float*)alloc((size_t)T_ * E_ * 4);
    float* gate   = (float*)alloc((size_t)T_ * 4);
    int*   eidx   = (int*)  alloc((size_t)T_ * 4);
    int*   pos    = (int*)  alloc((size_t)T_ * 4);
    f16*   buf16  = (f16*)  alloc((size_t)E_ * CAP_ * D_ * 2);
    f16*   w1t    = (f16*)  alloc((size_t)E_ * F_ * D_ * 2);
    f16*   w2t    = (f16*)  alloc((size_t)E_ * D_ * F_ * 2);
    f16*   h16    = (f16*)  alloc((size_t)E_ * CAP_ * F_ * 2);
    float* ybuf   = (float*)alloc((size_t)E_ * CAP_ * D_ * 4);

    float* out      = (float*)d_out;
    float* loss_out = out + (size_t)T_ * D_;

    int n;
    // pack / convert
    n = T_ * D_;      cvt_f16<<<(n + 255) / 256, 256, 0, stream>>>(src, src16, n);
    n = TD3_ * D_;    cvt_f16<<<(n + 255) / 256, 256, 0, stream>>>(inw, inw16, n);
    n = D_ * D_;      cvt_f16<<<(n + 255) / 256, 256, 0, stream>>>(outw, outw16, n);
    n = E_ * D_ * F_; tr_w1<<<(n + 255) / 256, 256, 0, stream>>>(w1, w1t, n);
    n = E_ * F_ * D_; tr_w2<<<(n + 255) / 256, 256, 0, stream>>>(w2, w2t, n);

    // QKV projection
    gemm_nt<float, false><<<dim3(TD3_ / 64, T_ / 128, 1), dim3(32, 8), 0, stream>>>(
        src16, inw16, inb, qkv, T_, TD3_, D_, 0, 0, 0, 0);

    n = T_ * D_;
    head_split<<<(n + 255) / 256, 256, 0, stream>>>(qkv, Qh, Kh, Vt);

    // attention (flash, online softmax)
    flash_attn<<<dim3(B_ * H_, S_ / 128), dim3(32, 8), 0, stream>>>(Qh, Kh, Vt, ctx16);

    // output projection
    gemm_nt<float, false><<<dim3(D_ / 64, T_ / 128, 1), dim3(32, 8), 0, stream>>>(
        ctx16, outw16, outb, attn, T_, D_, D_, 0, 0, 0, 0);

    // x = LN1(src + attn)
    ln_add<<<T_ / 8, dim3(32, 8), 0, stream>>>(src, attn, n1g, n1b, x, x16);

    // routing
    router_kernel<<<T_ / 128, 128, 0, stream>>>(x, rw, rb, probs, eidx, gate);
    expert_pos<<<1, 32, 0, stream>>>(eidx, pos);

    // dispatch
    n = E_ * CAP_ * D_;
    zero16<<<(n + 255) / 256, 256, 0, stream>>>(buf16, n);
    scatter_buf<<<T_, 256, 0, stream>>>(x16, eidx, pos, buf16);

    // expert FFN (batched over experts via blockIdx.z)
    gemm_nt<f16, true><<<dim3(F_ / 64, CAP_ / 128, E_), dim3(32, 8), 0, stream>>>(
        buf16, w1t, b1, h16, CAP_, F_, D_,
        (long long)CAP_ * D_, (long long)F_ * D_, (long long)F_, (long long)CAP_ * F_);
    gemm_nt<float, false><<<dim3(D_ / 64, CAP_ / 128, E_), dim3(32, 8), 0, stream>>>(
        h16, w2t, b2, ybuf, CAP_, D_, F_,
        (long long)CAP_ * F_, (long long)D_ * F_, (long long)D_, (long long)CAP_ * D_);

    // combine + LN2 -> out, then lb loss
    ln2_final<<<T_ / 8, dim3(32, 8), 0, stream>>>(x, ybuf, eidx, pos, gate, n2g, n2b, out);
    moe_loss<<<1, 256, 0, stream>>>(probs, eidx, loss_out);
}
